// DGCNNPointwiseNet_58952721105075
// MI455X (gfx1250) — matrix operand order repaired and checked by reading, not compiled, not run on hardware
//
#include <hip/hip_runtime.h>
#include <hip/hip_bf16.h>

typedef __attribute__((ext_vector_type(16))) _Float16 v16h;
typedef __attribute__((ext_vector_type(8)))  float    v8f;

#define SLOPE 0.2f
#define NB 8          // batches
#define NN 2048       // points per batch
#define MTOT (NB*NN)  // 16384 nodes
#define KNN 8
#define CTXD 515

__device__ __forceinline__ float lrelu(float v) { return v > 0.f ? v : SLOPE * v; }

// CDNA5 async global->LDS staging (ASYNCcnt-tracked, no VGPR round-trip).
// Guarded so the file still compiles (with sync staging) if the toolchain
// does not declare the gfx1250 async builtins.
#if defined(__gfx1250__) && __has_builtin(__builtin_amdgcn_global_load_async_to_lds_b128) && __has_builtin(__builtin_amdgcn_s_wait_asynccnt)
#define USE_ASYNC_LDS 1
#else
#define USE_ASYNC_LDS 0
#endif

#if USE_ASYNC_LDS
typedef int v4i32 __attribute__((vector_size(4 * sizeof(int))));
typedef __attribute__((address_space(1))) v4i32* gv4p;
typedef __attribute__((address_space(3))) v4i32* lv4p;
__device__ __forceinline__ void async_copy16(const void* g, void* l) {
  __builtin_amdgcn_global_load_async_to_lds_b128((gv4p)(void*)g, (lv4p)l, 0, 0);
}
#endif

// ---------------------------------------------------------------------------
// 0. ctx = [beta, sin(beta), cos(beta), context, cond]  -> (8, 515) f32
// ---------------------------------------------------------------------------
__global__ void build_ctx(const float* __restrict__ beta, const float* __restrict__ context,
                          const float* __restrict__ cond, float* __restrict__ ctxf) {
  int id = blockIdx.x * blockDim.x + threadIdx.x;
  if (id >= NB * CTXD) return;
  int b = id / CTXD, j = id % CTXD;
  float v;
  if (j == 0)        v = beta[b];
  else if (j == 1)   v = sinf(beta[b]);
  else if (j == 2)   v = cosf(beta[b]);
  else if (j < 259)  v = context[b * 256 + (j - 3)];
  else               v = cond[b * 256 + (j - 259)];
  ctxf[id] = v;
}

// ---------------------------------------------------------------------------
// 1. kNN (K=8) per batch, self excluded, fp32 (matches reference formula)
//    grid = (NN/256, NB), block = 256; batch points staged in LDS
// ---------------------------------------------------------------------------
__global__ void __launch_bounds__(256) knn_kernel(const float* __restrict__ x, int* __restrict__ idx) {
  __shared__ float xs[NN * 3];
  __shared__ float sq[NN];
  const int b = blockIdx.y;
  const int t = threadIdx.x;
  const float* xb = x + (size_t)b * NN * 3;
  for (int i = t; i < NN; i += 256) {
    float x0 = xb[i * 3], x1 = xb[i * 3 + 1], x2 = xb[i * 3 + 2];
    xs[i * 3] = x0; xs[i * 3 + 1] = x1; xs[i * 3 + 2] = x2;
    sq[i] = x0 * x0 + x1 * x1 + x2 * x2;
  }
  __syncthreads();
  const int q = blockIdx.x * 256 + t;
  const float qx = xs[q * 3], qy = xs[q * 3 + 1], qz = xs[q * 3 + 2], sqq = sq[q];
  float bd[KNN]; int bi[KNN];
  for (int k = 0; k < KNN; ++k) { bd[k] = 3.0e38f; bi[k] = 0; }
  for (int m = 0; m < NN; ++m) {
    if (m == q) continue;
    float d = sqq + sq[m] - 2.0f * (qx * xs[m * 3] + qy * xs[m * 3 + 1] + qz * xs[m * 3 + 2]);
    if (d < bd[KNN - 1]) {
      int j = KNN - 1;
      while (j > 0 && d < bd[j - 1]) { bd[j] = bd[j - 1]; bi[j] = bi[j - 1]; --j; }
      bd[j] = d; bi[j] = m;
    }
  }
  int* op = idx + ((size_t)b * NN + q) * KNN;
  for (int k = 0; k < KNN; ++k) op[k] = bi[k];
}

// ---------------------------------------------------------------------------
// 2. init MLP: 3->64 linear, leaky, LayerNorm(64); 64 threads per node
// ---------------------------------------------------------------------------
__global__ void __launch_bounds__(256) init_mlp(const float* __restrict__ x, const float* __restrict__ W0,
                                                const float* __restrict__ b0, const float* __restrict__ g,
                                                const float* __restrict__ bb, _Float16* __restrict__ feat) {
  __shared__ float red[256];
  const int t = threadIdx.x;
  const int node = blockIdx.x * 4 + (t >> 6);
  const int c = t & 63;
  const float x0 = x[node * 3], x1 = x[node * 3 + 1], x2 = x[node * 3 + 2];
  float h = x0 * W0[c] + x1 * W0[64 + c] + x2 * W0[128 + c] + b0[c];
  h = lrelu(h);
  red[t] = h; __syncthreads();
  for (int s = 32; s >= 1; s >>= 1) { if (c < s) red[t] += red[t + s]; __syncthreads(); }
  const float mu = red[t & ~63] * (1.f / 64.f);
  __syncthreads();
  const float dv = h - mu;
  red[t] = dv * dv; __syncthreads();
  for (int s = 32; s >= 1; s >>= 1) { if (c < s) red[t] += red[t + s]; __syncthreads(); }
  const float var = red[t & ~63] * (1.f / 64.f);
  const float f = dv * rsqrtf(var + 1e-5f) * g[c] + bb[c];
  feat[(size_t)node * 64 + c] = (_Float16)f;
}

// ---------------------------------------------------------------------------
// 3a. fold edge-conv weights: Bt(2*out x D), row r<out: W[k][r]-W[D+k][r]
//     (center minus neighbor); row r>=out: W[D+k][r-out] (neighbor). N-major.
// ---------------------------------------------------------------------------
__global__ void prep_dualw(const float* __restrict__ W, _Float16* __restrict__ Bt, int D, int out) {
  int id = blockIdx.x * blockDim.x + threadIdx.x;
  if (id >= 2 * out * D) return;
  int r = id / D, k = id % D;
  float v;
  if (r < out) v = W[(size_t)k * out + r] - W[(size_t)(D + k) * out + r];
  else         v = W[(size_t)(D + k) * out + (r - out)];
  Bt[id] = (_Float16)v;
}

// 3b. plain transpose to f16 (for Wf1): Bt[n*K+k] = W[k*N+n]
__global__ void prep_transpose(const float* __restrict__ W, _Float16* __restrict__ Bt, int K, int N) {
  int id = blockIdx.x * blockDim.x + threadIdx.x;
  if (id >= N * K) return;
  int n = id / K, k = id % K;
  Bt[id] = (_Float16)W[(size_t)k * N + n];
}

// 3c. per-batch ctx contribution + bias: ctxb[b][n] = b[n] + ctx[b]·W[2D: , n]
__global__ void ctxb_kernel(const float* __restrict__ W, const float* __restrict__ bias,
                            const float* __restrict__ ctxf, float* __restrict__ ctxb, int D, int out) {
  const int n = blockIdx.x * blockDim.x + threadIdx.x;
  const int b = blockIdx.y;
  if (n >= out) return;
  const float* Wc = W + (size_t)2 * D * out;
  const float* cf = ctxf + b * CTXD;
  float acc = bias[n];
  for (int j = 0; j < CTXD; ++j) acc += cf[j] * Wc[(size_t)j * out + n];
  ctxb[b * out + n] = acc;
}

// ---------------------------------------------------------------------------
// 4. f16 WMMA GEMM: C(MxN f32) = A(MxK f16, row-major) * Bt(NxK f16, N-major)^T
//    Tile 128x128x32, 8 waves (4x2), wave tile 32x64 (2x4 accumulators).
//    Fragment layouts per CDNA5 ISA 7.12.2. Tile staging uses
//    GLOBAL_LOAD_ASYNC_TO_LDS_B128 when available (ASYNCcnt), else
//    global_load_b128 + ds_store_b128.
// ---------------------------------------------------------------------------
#define BM 128
#define BN 128
#define BKK 32

union FragU { v16h v; uint4 u[2]; };

__global__ void __launch_bounds__(256) gemm_f16_wmma(const _Float16* __restrict__ A,
                                                     const _Float16* __restrict__ Bt,
                                                     float* __restrict__ C, int M, int N, int K) {
  __shared__ _Float16 As[BM][BKK];
  __shared__ _Float16 Bs[BN][BKK];
  const int t = threadIdx.x;
  const int wid = t >> 5, lane = t & 31;
  const int half = lane >> 4, r = lane & 15;
  const int rowBase = blockIdx.x * BM;
  const int colBase = blockIdx.y * BN;
  const int wm = (wid & 3) * 32;   // wave row offset inside tile
  const int wn = (wid >> 2) * 64;  // wave col offset inside tile

  v8f zero = {};
  v8f acc[2][4];
  for (int i = 0; i < 2; ++i)
    for (int j = 0; j < 4; ++j) acc[i][j] = zero;

  const int nk = K / BKK;
  for (int kt = 0; kt < nk; ++kt) {
    const int k0 = kt * BKK;
    __syncthreads();
    // stage tiles: 128 rows x 32 f16 = 512 x 16B each; 2 x 16B per thread
    for (int p = 0; p < 2; ++p) {
      int u = t + p * 256;
      int row = u >> 2, c4 = u & 3;
#if USE_ASYNC_LDS
      async_copy16(&A[(size_t)(rowBase + row) * K + k0 + c4 * 8], &As[row][c4 * 8]);
      async_copy16(&Bt[(size_t)(colBase + row) * K + k0 + c4 * 8], &Bs[row][c4 * 8]);
#else
      *(uint4*)&As[row][c4 * 8] = *(const uint4*)&A[(size_t)(rowBase + row) * K + k0 + c4 * 8];
      *(uint4*)&Bs[row][c4 * 8] = *(const uint4*)&Bt[(size_t)(colBase + row) * K + k0 + c4 * 8];
#endif
    }
    if (kt + 1 < nk) {  // speculative prefetch of next A k-slab (near-temporal)
      __builtin_prefetch(&A[(size_t)(rowBase + (t >> 1)) * K + k0 + BKK], 0, 3);
    }
#if USE_ASYNC_LDS
    __builtin_amdgcn_s_wait_asynccnt(0);  // own async writes done before barrier
#endif
    __syncthreads();

    v16h afr[2], bfr[4];
    for (int i = 0; i < 2; ++i) {   // A 16x32: chunk0 K=half*8, chunk1 K=16+half*8
      FragU fa;
      const int mrow = wm + i * 16 + r;
      fa.u[0] = *(const uint4*)&As[mrow][half * 8];
      fa.u[1] = *(const uint4*)&As[mrow][16 + half * 8];
      afr[i] = fa.v;
    }
    for (int j = 0; j < 4; ++j) {   // B 32x16: chunk0 K=half*8, chunk1 K=16+half*8
      FragU fb;
      const int nrow = wn + j * 16 + r;
      fb.u[0] = *(const uint4*)&Bs[nrow][half * 8];
      fb.u[1] = *(const uint4*)&Bs[nrow][16 + half * 8];
      bfr[j] = fb.v;
    }
    for (int i = 0; i < 2; ++i)
      for (int j = 0; j < 4; ++j)
        acc[i][j] = __builtin_amdgcn_wmma_f32_16x16x32_f16(
            false, afr[i], false, bfr[j], (short)0, acc[i][j], false, false);
  }
  // C/D layout: VGPR v -> M = v + half*8, N = r
  for (int i = 0; i < 2; ++i)
    for (int j = 0; j < 4; ++j)
      for (int v = 0; v < 8; ++v) {
        const int mrow = rowBase + wm + i * 16 + v + half * 8;
        const int ncol = colBase + wn + j * 16 + r;
        C[(size_t)mrow * N + ncol] = acc[i][j][v];
      }
}

// ---------------------------------------------------------------------------
// 5. edge max: out[n,c] = max_k lrelu( C_part[n,c] + Z_part[idx,c] + ctxb[b,c] )
//    CZ is (M x 2*out): cols [0,out) = center part, [out,2out) = neighbor part
// ---------------------------------------------------------------------------
__global__ void edge_max(const float* __restrict__ CZ, const int* __restrict__ idx,
                         const float* __restrict__ ctxb, _Float16* __restrict__ featOut, int out) {
  __shared__ int sidx[KNN];
  const int n = blockIdx.x;
  const int b = n >> 11;
  const int t = threadIdx.x;
  if (t < KNN) sidx[t] = idx[(size_t)n * KNN + t];
  __syncthreads();
  const int N2 = out * 2;
  const float base = CZ[(size_t)n * N2 + t] + ctxb[b * out + t];
  float best = -3.0e38f;
  for (int k = 0; k < KNN; ++k) {
    const int mg = (b << 11) + sidx[k];
    float v = base + CZ[(size_t)mg * N2 + out + t];
    best = fmaxf(best, lrelu(v));
  }
  featOut[(size_t)n * out + t] = (_Float16)best;
}

// ---------------------------------------------------------------------------
// 6. final: out = x + lrelu(a1 + bf1) @ Wf2 + bf2   (a1 = x3@Wf1 from GEMM)
// ---------------------------------------------------------------------------
__global__ void __launch_bounds__(256) final_kernel(const float* __restrict__ A1,
                                                    const float* __restrict__ bf1,
                                                    const float* __restrict__ Wf2,
                                                    const float* __restrict__ bf2,
                                                    const float* __restrict__ x,
                                                    float* __restrict__ out) {
  __shared__ float sW[256 * 3];
  __shared__ float sb[256];
  const int t = threadIdx.x;
  for (int i = t; i < 768; i += 256) sW[i] = Wf2[i];
  sb[t] = bf1[t];
  __syncthreads();
  const int n = blockIdx.x * 256 + t;
  const float* a = A1 + (size_t)n * 256;
  float acc0 = bf2[0], acc1 = bf2[1], acc2 = bf2[2];
  for (int c = 0; c < 256; ++c) {
    float v = lrelu(a[c] + sb[c]);
    acc0 += v * sW[c * 3];
    acc1 += v * sW[c * 3 + 1];
    acc2 += v * sW[c * 3 + 2];
  }
  out[n * 3]     = x[n * 3]     + acc0;
  out[n * 3 + 1] = x[n * 3 + 1] + acc1;
  out[n * 3 + 2] = x[n * 3 + 2] + acc2;
}

// ---------------------------------------------------------------------------
extern "C" void kernel_launch(void* const* d_in, const int* in_sizes, int n_in,
                              void* d_out, int out_size, void* d_ws, size_t ws_size,
                              hipStream_t stream) {
  (void)in_sizes; (void)n_in; (void)out_size; (void)ws_size;
  const float* x       = (const float*)d_in[0];
  const float* beta    = (const float*)d_in[1];
  const float* context = (const float*)d_in[2];
  const float* cond    = (const float*)d_in[3];
  const float* W0      = (const float*)d_in[4];
  const float* b0      = (const float*)d_in[5];
  const float* ln_g    = (const float*)d_in[6];
  const float* ln_b    = (const float*)d_in[7];
  const float* W1      = (const float*)d_in[8];
  const float* b1      = (const float*)d_in[9];
  const float* W2      = (const float*)d_in[10];
  const float* b2      = (const float*)d_in[11];
  const float* W3      = (const float*)d_in[12];
  const float* b3      = (const float*)d_in[13];
  const float* Wf1     = (const float*)d_in[14];
  const float* bf1     = (const float*)d_in[15];
  const float* Wf2     = (const float*)d_in[16];
  const float* bf2     = (const float*)d_in[17];
  float* out = (float*)d_out;

  // workspace carve (256B aligned)
  char* ws = (char*)d_ws;
  auto carve = [&](size_t bytes) -> void* {
    void* p = (void*)ws;
    ws += (bytes + 255) & ~(size_t)255;
    return p;
  };
  float*     ctxf  = (float*)    carve((size_t)NB * CTXD * 4);
  int*       idx   = (int*)      carve((size_t)MTOT * KNN * 4);
  _Float16*  feat0 = (_Float16*) carve((size_t)MTOT * 64 * 2);
  _Float16*  featA = (_Float16*) carve((size_t)MTOT * 512 * 2);  // x1, then x3
  _Float16*  featB = (_Float16*) carve((size_t)MTOT * 512 * 2);  // x2
  _Float16*  wl1   = (_Float16*) carve((size_t)256 * 64 * 2);
  _Float16*  wl2   = (_Float16*) carve((size_t)512 * 128 * 2);
  _Float16*  wl3   = (_Float16*) carve((size_t)1024 * 256 * 2);
  _Float16*  wf1t  = (_Float16*) carve((size_t)256 * 512 * 2);
  float*     ctxb1 = (float*)    carve((size_t)NB * 128 * 4);
  float*     ctxb2 = (float*)    carve((size_t)NB * 256 * 4);
  float*     ctxb3 = (float*)    carve((size_t)NB * 512 * 4);
  float*     CZ    = (float*)    carve((size_t)MTOT * 1024 * 4); // reused each layer

  // stage 0: context vector + kNN + init MLP
  build_ctx<<<(NB * CTXD + 255) / 256, 256, 0, stream>>>(beta, context, cond, ctxf);
  knn_kernel<<<dim3(NN / 256, NB), 256, 0, stream>>>(x, idx);
  init_mlp<<<MTOT / 4, 256, 0, stream>>>(x, W0, b0, ln_g, ln_b, feat0);

  // weight prep
  prep_dualw<<<(2 * 128 * 64 + 255) / 256, 256, 0, stream>>>(W1, wl1, 64, 128);
  prep_dualw<<<(2 * 256 * 128 + 255) / 256, 256, 0, stream>>>(W2, wl2, 128, 256);
  prep_dualw<<<(2 * 512 * 256 + 255) / 256, 256, 0, stream>>>(W3, wl3, 256, 512);
  prep_transpose<<<(256 * 512 + 255) / 256, 256, 0, stream>>>(Wf1, wf1t, 512, 256);
  ctxb_kernel<<<dim3(2, NB), 64, 0, stream>>>(W1, b1, ctxf, ctxb1, 64, 128);
  ctxb_kernel<<<dim3(4, NB), 64, 0, stream>>>(W2, b2, ctxf, ctxb2, 128, 256);
  ctxb_kernel<<<dim3(8, NB), 64, 0, stream>>>(W3, b3, ctxf, ctxb3, 256, 512);

  // edge conv 1: (16384x64) x (64x256) -> CZ, then gather-max -> x1 (128)
  gemm_f16_wmma<<<dim3(MTOT / BM, 256 / BN), 256, 0, stream>>>(feat0, wl1, CZ, MTOT, 256, 64);
  edge_max<<<MTOT, 128, 0, stream>>>(CZ, idx, ctxb1, featA, 128);

  // edge conv 2: (16384x128) x (128x512) -> CZ, gather-max -> x2 (256)
  gemm_f16_wmma<<<dim3(MTOT / BM, 512 / BN), 256, 0, stream>>>(featA, wl2, CZ, MTOT, 512, 128);
  edge_max<<<MTOT, 256, 0, stream>>>(CZ, idx, ctxb2, featB, 256);

  // edge conv 3: (16384x256) x (256x1024) -> CZ, gather-max -> x3 (512)
  gemm_f16_wmma<<<dim3(MTOT / BM, 1024 / BN), 256, 0, stream>>>(featB, wl3, CZ, MTOT, 1024, 256);
  edge_max<<<MTOT, 512, 0, stream>>>(CZ, idx, ctxb3, featA, 512);

  // final MLP: a1 = x3 @ Wf1 (16384x256), then 256->3 + residual
  gemm_f16_wmma<<<dim3(MTOT / BM, 256 / BN), 256, 0, stream>>>(featA, wf1t, CZ, MTOT, 256, 512);
  final_kernel<<<MTOT / 256, 256, 0, stream>>>(CZ, bf1, Wf2, bf2, x, out);
}
